// MegaBlockFeedForward_57097295233529
// MI455X (gfx1250) — matrix operand order repaired and checked by reading, not compiled, not run on hardware
//
#include <hip/hip_runtime.h>
#include <cstdint>

// Grouped SwiGLU FFN for MI455X (gfx1250, wave32, WMMA, async global->LDS).
// Full path (ws >= 480MB):
//   P0: convert x, w3, w2 to bf16; convert+transpose w1 -> [E,H,D] bf16
//   K1: h = silu(x@w1) * (x@w3^T) -> bf16 [E,T,H]   (pure-copy staging, async)
//   K2: out = h @ w2^T            -> f32 [E,T,D]
// Fallback path: in-loop f32->bf16 staging (no pre-convert).

#define E_ 8
#define T_ 1024
#define D_ 2048
#define H_ 4096

#if __has_builtin(__builtin_amdgcn_global_load_async_to_lds_b128)
#define HAVE_ASYNC_LDS 1
#else
#define HAVE_ASYNC_LDS 0
#endif

typedef __attribute__((ext_vector_type(16))) __bf16 v16bf;
typedef __attribute__((ext_vector_type(8)))  __bf16 v8bf;
typedef __attribute__((ext_vector_type(8)))  float  v8f;
typedef __attribute__((ext_vector_type(4)))  int    v4i;

static __device__ __forceinline__ uint16_t f2bf_bits(float f) {
  uint32_t u = __builtin_bit_cast(uint32_t, f);
  return (uint16_t)((u + 0x7FFFu + ((u >> 16) & 1u)) >> 16);
}
static __device__ __forceinline__ __bf16 f2bf(float f) {
  return __builtin_bit_cast(__bf16, f2bf_bits(f));
}

// 16-byte global -> LDS copy; async (ASYNCcnt) when the builtin exists.
static __device__ __forceinline__ void cp16_g2l(const void* g, void* l) {
#if HAVE_ASYNC_LDS
  typedef __attribute__((address_space(1))) v4i* gp_t;  // prints as __device__*
  typedef __attribute__((address_space(3))) v4i* lp_t;  // prints as __shared__*
  __builtin_amdgcn_global_load_async_to_lds_b128((gp_t)g, (lp_t)l, 0, 0);
#else
  *(v8bf*)l = *(const v8bf*)g;
#endif
}
static __device__ __forceinline__ void wait_cp() {
#if HAVE_ASYNC_LDS
#if __has_builtin(__builtin_amdgcn_s_wait_asynccnt)
  __builtin_amdgcn_s_wait_asynccnt(0);
#else
  asm volatile("s_wait_asynccnt 0" ::: "memory");
#endif
#endif
}

static __device__ __forceinline__ v16bf combine8(v8bf lo, v8bf hi) {
  v16bf a;
#pragma unroll
  for (int i = 0; i < 8; ++i) { a[i] = lo[i]; a[i + 8] = hi[i]; }
  return a;
}

// ---------------------------------------------------------------------------
// P0a: elementwise f32 -> bf16 (4 elems/thread)
// ---------------------------------------------------------------------------
__global__ __launch_bounds__(256, 1)
void cvt_f32_bf16_kernel(const float* __restrict__ in,
                         unsigned short* __restrict__ out) {
  size_t i = ((size_t)blockIdx.x * 256 + threadIdx.x) * 4;
  float4 v = *(const float4*)(in + i);
  uint2 p;
  p.x = (uint32_t)f2bf_bits(v.x) | ((uint32_t)f2bf_bits(v.y) << 16);
  p.y = (uint32_t)f2bf_bits(v.z) | ((uint32_t)f2bf_bits(v.w) << 16);
  *(uint2*)(out + i) = p;
}

// ---------------------------------------------------------------------------
// P0b: w1 [E,D,H] f32 -> [E,H,D] bf16 (32x32 LDS transpose tiles)
// ---------------------------------------------------------------------------
__global__ __launch_bounds__(256, 1)
void transpose_cvt_kernel(const float* __restrict__ in,
                          unsigned short* __restrict__ out) {
  __shared__ float tile[32][33];
  const int e  = blockIdx.z;
  const int h0 = blockIdx.x * 32, d0 = blockIdx.y * 32;
  const float* src = in + (size_t)e * D_ * H_;
  unsigned short* dst = out + (size_t)e * D_ * H_;
#pragma unroll
  for (int i = 0; i < 4; ++i) {
    int d = d0 + threadIdx.y + i * 8;
    tile[threadIdx.y + i * 8][threadIdx.x] = src[(size_t)d * H_ + h0 + threadIdx.x];
  }
  __syncthreads();
#pragma unroll
  for (int i = 0; i < 4; ++i) {
    int h = h0 + threadIdx.y + i * 8;
    dst[(size_t)h * D_ + d0 + threadIdx.x] =
        f2bf_bits(tile[threadIdx.x][threadIdx.y + i * 8]);
  }
}

// ---------------------------------------------------------------------------
// K1: block tile 128(M) x 64(N), K-step 32 over D, double-buffered LDS.
// 8 waves 4(M) x 2(N); wave tile 32x32 -> 2x2 frags, two acc sets.
// PRE: inputs are bf16 (x [E,T,D]; w1t/w3 [E,H,D]) -> pure-copy staging.
// ---------------------------------------------------------------------------
template <bool PRE>
__global__ __launch_bounds__(256, 1)
void swiglu_up_gate_kernel(const void* __restrict__ xin,
                           const void* __restrict__ b1in,
                           const void* __restrict__ b3in,
                           unsigned short* __restrict__ hbuf_raw) {
  __bf16* hbuf = (__bf16*)hbuf_raw;
  const int e   = blockIdx.z;
  const int m0  = blockIdx.x * 128;
  const int n0  = blockIdx.y * 64;
  const int tid = threadIdx.x;

  __shared__ __align__(16) __bf16 sA [2][128 * 32];
  __shared__ __align__(16) __bf16 sB1[2][ 64 * 32];
  __shared__ __align__(16) __bf16 sB3[2][ 64 * 32];

  const int wave = tid >> 5, lane = tid & 31;
  const int wm   = (wave >> 1) * 32;
  const int wn   = (wave & 1) * 32;
  const int half = lane >> 4;
  const int r    = lane & 15;

  v8f acc1[2][2] = {};
  v8f acc3[2][2] = {};

  auto stage = [&](int k0, int buf) {
    if (PRE) {
      const __bf16* xbf = (const __bf16*)xin;
      const __bf16* w1t = (const __bf16*)b1in;  // [E,H,D]
      const __bf16* w3b = (const __bf16*)b3in;  // [E,H,D]
      {
        const int row = tid >> 1, kk = (tid & 1) * 16;
        const __bf16* g = xbf + (size_t)(e * T_ + m0 + row) * D_ + k0 + kk;
        cp16_g2l(g,     &sA[buf][row * 32 + kk]);
        cp16_g2l(g + 8, &sA[buf][row * 32 + kk + 8]);
      }
      {
        const int nrow = tid >> 2, kk = (tid & 3) * 8;
        const size_t off = (size_t)(e * H_ + n0 + nrow) * D_ + k0 + kk;
        cp16_g2l(w1t + off, &sB1[buf][nrow * 32 + kk]);
        cp16_g2l(w3b + off, &sB3[buf][nrow * 32 + kk]);
      }
    } else {
      const float* x  = (const float*)xin;
      const float* w1 = (const float*)b1in;  // [E,D,H]
      const float* w3 = (const float*)b3in;  // [E,H,D]
      {
        const int row = tid >> 1, kk = (tid & 1) * 16;
        const float* g = x + (size_t)(e * T_ + m0 + row) * D_ + k0 + kk;
#pragma unroll
        for (int j = 0; j < 16; j += 4) {
          float4 v = *(const float4*)(g + j);
          __bf16* s = &sA[buf][row * 32 + kk + j];
          s[0] = f2bf(v.x); s[1] = f2bf(v.y); s[2] = f2bf(v.z); s[3] = f2bf(v.w);
        }
      }
      {
        const int krow = tid >> 3, nn = (tid & 7) * 8;
        const float* g = w1 + (size_t)(e * D_ + k0 + krow) * H_ + n0 + nn;
#pragma unroll
        for (int j = 0; j < 8; j += 4) {
          float4 v = *(const float4*)(g + j);
          sB1[buf][(nn + j + 0) * 32 + krow] = f2bf(v.x);
          sB1[buf][(nn + j + 1) * 32 + krow] = f2bf(v.y);
          sB1[buf][(nn + j + 2) * 32 + krow] = f2bf(v.z);
          sB1[buf][(nn + j + 3) * 32 + krow] = f2bf(v.w);
        }
      }
      {
        const int nrow = tid >> 2, kk = (tid & 3) * 8;
        const float* g = w3 + (size_t)(e * H_ + n0 + nrow) * D_ + k0 + kk;
#pragma unroll
        for (int j = 0; j < 8; j += 4) {
          float4 v = *(const float4*)(g + j);
          __bf16* s = &sB3[buf][nrow * 32 + kk + j];
          s[0] = f2bf(v.x); s[1] = f2bf(v.y); s[2] = f2bf(v.z); s[3] = f2bf(v.w);
        }
      }
    }
  };

  auto compute = [&](int buf) {
    v16bf a[2], b1f[2], b3f[2];
#pragma unroll
    for (int mi = 0; mi < 2; ++mi) {
      const __bf16* p = &sA[buf][(wm + mi * 16 + r) * 32 + half * 16];
      a[mi] = combine8(*(const v8bf*)p, *(const v8bf*)(p + 8));
    }
#pragma unroll
    for (int ni = 0; ni < 2; ++ni) {
      const __bf16* p1 = &sB1[buf][(wn + ni * 16 + r) * 32 + half * 16];
      b1f[ni] = combine8(*(const v8bf*)p1, *(const v8bf*)(p1 + 8));
      const __bf16* p3 = &sB3[buf][(wn + ni * 16 + r) * 32 + half * 16];
      b3f[ni] = combine8(*(const v8bf*)p3, *(const v8bf*)(p3 + 8));
    }
#pragma unroll
    for (int mi = 0; mi < 2; ++mi)
#pragma unroll
      for (int ni = 0; ni < 2; ++ni) {
        acc1[mi][ni] = __builtin_amdgcn_wmma_f32_16x16x32_bf16(
            false, a[mi], false, b1f[ni], (short)0, acc1[mi][ni], false, false);
        acc3[mi][ni] = __builtin_amdgcn_wmma_f32_16x16x32_bf16(
            false, a[mi], false, b3f[ni], (short)0, acc3[mi][ni], false, false);
      }
  };

  const int NK = D_ / 32;
  stage(0, 0);
  wait_cp();
  __syncthreads();
  for (int kt = 0; kt < NK; ++kt) {
    const int cur = kt & 1;
    if (kt + 1 < NK) stage((kt + 1) * 32, cur ^ 1);  // overlap with WMMA
    compute(cur);
    wait_cp();
    __syncthreads();
  }

  // SiLU gate epilogue (hardware rcp, no IEEE div chain)
#pragma unroll
  for (int mi = 0; mi < 2; ++mi)
#pragma unroll
    for (int ni = 0; ni < 2; ++ni) {
      const int col = n0 + wn + ni * 16 + r;
#pragma unroll
      for (int v = 0; v < 8; ++v) {
        const int row = m0 + wm + mi * 16 + half * 8 + v;
        float g   = acc1[mi][ni][v];
        float sig = __builtin_amdgcn_rcpf(1.0f + __expf(-g));
        hbuf[(size_t)(e * T_ + row) * H_ + col] =
            f2bf(g * sig * acc3[mi][ni][v]);
      }
    }
}

// ---------------------------------------------------------------------------
// K2: out = h @ w2^T. Block tile 128x128, K-step 32 over H, double-buffered.
// 8 waves 4(M) x 2(N); wave tile 32x64 -> 2x4 frags.
// ---------------------------------------------------------------------------
template <bool PRE>
__global__ __launch_bounds__(256, 1)
void down_proj_kernel(const unsigned short* __restrict__ hbuf_raw,
                      const void* __restrict__ w2in,
                      float* __restrict__ out) {
  const __bf16* hbuf = (const __bf16*)hbuf_raw;
  const int e   = blockIdx.z;
  const int m0  = blockIdx.x * 128;
  const int n0  = blockIdx.y * 128;
  const int tid = threadIdx.x;

  __shared__ __align__(16) __bf16 sA[2][128 * 32];
  __shared__ __align__(16) __bf16 sB[2][128 * 32];

  const int wave = tid >> 5, lane = tid & 31;
  const int wm   = (wave >> 1) * 32;
  const int wn   = (wave & 1) * 64;
  const int half = lane >> 4;
  const int r    = lane & 15;

  v8f acc[2][4] = {};

  auto stage = [&](int k0, int buf) {
    {
      const int row = tid >> 1, kk = (tid & 1) * 16;
      const __bf16* g = hbuf + (size_t)(e * T_ + m0 + row) * H_ + k0 + kk;
      cp16_g2l(g,     &sA[buf][row * 32 + kk]);
      cp16_g2l(g + 8, &sA[buf][row * 32 + kk + 8]);
    }
    if (PRE) {
      const __bf16* w2b = (const __bf16*)w2in;  // [E,D,H] bf16
      const int nrow = tid >> 1, kk = (tid & 1) * 16;
      const __bf16* g = w2b + (size_t)(e * D_ + n0 + nrow) * H_ + k0 + kk;
      cp16_g2l(g,     &sB[buf][nrow * 32 + kk]);
      cp16_g2l(g + 8, &sB[buf][nrow * 32 + kk + 8]);
    } else {
      const float* w2 = (const float*)w2in;
      const int nrow = tid >> 1, kk = (tid & 1) * 16;
      const float* g = w2 + (size_t)(e * D_ + n0 + nrow) * H_ + k0 + kk;
#pragma unroll
      for (int j = 0; j < 16; j += 4) {
        float4 v = *(const float4*)(g + j);
        __bf16* s = &sB[buf][nrow * 32 + kk + j];
        s[0] = f2bf(v.x); s[1] = f2bf(v.y); s[2] = f2bf(v.z); s[3] = f2bf(v.w);
      }
    }
  };

  auto compute = [&](int buf) {
    v16bf a[2], b[4];
#pragma unroll
    for (int mi = 0; mi < 2; ++mi) {
      const __bf16* p = &sA[buf][(wm + mi * 16 + r) * 32 + half * 16];
      a[mi] = combine8(*(const v8bf*)p, *(const v8bf*)(p + 8));
    }
#pragma unroll
    for (int ni = 0; ni < 4; ++ni) {
      const __bf16* p = &sB[buf][(wn + ni * 16 + r) * 32 + half * 16];
      b[ni] = combine8(*(const v8bf*)p, *(const v8bf*)(p + 8));
    }
#pragma unroll
    for (int mi = 0; mi < 2; ++mi)
#pragma unroll
      for (int ni = 0; ni < 4; ++ni)
        acc[mi][ni] = __builtin_amdgcn_wmma_f32_16x16x32_bf16(
            false, a[mi], false, b[ni], (short)0, acc[mi][ni], false, false);
  };

  const int NK = H_ / 32;
  stage(0, 0);
  wait_cp();
  __syncthreads();
  for (int kt = 0; kt < NK; ++kt) {
    const int cur = kt & 1;
    if (kt + 1 < NK) stage((kt + 1) * 32, cur ^ 1);
    compute(cur);
    wait_cp();
    __syncthreads();
  }

#pragma unroll
  for (int mi = 0; mi < 2; ++mi)
#pragma unroll
    for (int ni = 0; ni < 4; ++ni) {
      const int col = n0 + wn + ni * 16 + r;
#pragma unroll
      for (int v = 0; v < 8; ++v) {
        const int row = m0 + wm + mi * 16 + half * 8 + v;
        out[(size_t)(e * T_ + row) * D_ + col] = acc[mi][ni][v];
      }
    }
}

// ---------------------------------------------------------------------------
extern "C" void kernel_launch(void* const* d_in, const int* in_sizes, int n_in,
                              void* d_out, int out_size, void* d_ws, size_t ws_size,
                              hipStream_t stream) {
  (void)in_sizes; (void)n_in; (void)out_size;
  const float* x  = (const float*)d_in[0];
  const float* w1 = (const float*)d_in[1];
  const float* w2 = (const float*)d_in[2];
  const float* w3 = (const float*)d_in[3];
  float* out = (float*)d_out;

  const size_t hB = (size_t)E_ * T_ * H_ * 2;  //  64 MB  h  (bf16)
  const size_t xB = (size_t)E_ * T_ * D_ * 2;  //  32 MB  x  (bf16)
  const size_t wB = (size_t)E_ * D_ * H_ * 2;  // 128 MB  per weight (bf16)
  const bool full = ws_size >= hB + xB + 3 * wB;

  unsigned short* hbuf = (unsigned short*)d_ws;
  dim3 blk(256, 1, 1);
  dim3 g1(T_ / 128, H_ / 64, E_);   // 8 x 64 x 8
  dim3 g2(T_ / 128, D_ / 128, E_);  // 8 x 16 x 8

  if (full) {
    char* base = (char*)d_ws;
    unsigned short* xbf = (unsigned short*)(base + hB);
    unsigned short* w1t = (unsigned short*)(base + hB + xB);
    unsigned short* w3b = (unsigned short*)(base + hB + xB + wB);
    unsigned short* w2b = (unsigned short*)(base + hB + xB + 2 * wB);

    const size_t nx = (size_t)E_ * T_ * D_;  // 16M
    const size_t nw = (size_t)E_ * D_ * H_;  // 64M
    cvt_f32_bf16_kernel<<<(unsigned)(nx / 1024), blk, 0, stream>>>(x, xbf);
    cvt_f32_bf16_kernel<<<(unsigned)(nw / 1024), blk, 0, stream>>>(w3, w3b);
    cvt_f32_bf16_kernel<<<(unsigned)(nw / 1024), blk, 0, stream>>>(w2, w2b);
    transpose_cvt_kernel<<<dim3(H_ / 32, D_ / 32, E_), dim3(32, 8, 1), 0, stream>>>(w1, w1t);

    swiglu_up_gate_kernel<true><<<g1, blk, 0, stream>>>(xbf, w1t, w3b, hbuf);
    down_proj_kernel<true><<<g2, blk, 0, stream>>>(hbuf, w2b, out);
  } else {
    swiglu_up_gate_kernel<false><<<g1, blk, 0, stream>>>(x, w1, w3, hbuf);
    down_proj_kernel<false><<<g2, blk, 0, stream>>>(hbuf, w2, out);
  }
}
